// Multi_domain_FeedForwardNetwork_v9_87144886435924
// MI455X (gfx1250) — matrix operand order repaired and checked by reading, not compiled, not run on hardware
//
#include <hip/hip_runtime.h>

// MI455X / gfx1250, wave32. bf16 WMMA GEMM pipeline with async global->LDS
// double-buffered staging (ASYNCcnt) feeding v_wmma_f32_16x16x32_bf16.
// Domain mask is one-hot -> compute only the active domain per token.
//
// Workspace (~104 MiB):
//   xperm bf16 [8][512][1024], wti bf16 [12288][1024], wto bf16 [6][1024][2048],
//   hid f32 [8][512][2048], hbf bf16 [8][512][2048]

typedef __attribute__((ext_vector_type(16))) __bf16 v16bf;
typedef __attribute__((ext_vector_type(8)))  float  v8f;
typedef __attribute__((ext_vector_type(4)))  int    v4i;
typedef __attribute__((address_space(1))) v4i* gptr_b128;   // global int4*
typedef __attribute__((address_space(3))) v4i* lptr_b128;   // LDS int4*

#define BB    8
#define TT    512
#define DIN   1024
#define INNER 2048
#define DOUT  1024
#define NDOM  6
#define NTOK  (BB * TT)
#define KPG   (NTOK / BB)

__device__ __forceinline__ unsigned short f2bf(float f) {
  unsigned int u = __float_as_uint(f);
  u += 0x7FFFu + ((u >> 16) & 1u);
  return (unsigned short)(u >> 16);
}

__device__ __forceinline__ float block_reduce_sum(float v, float* red) {
  int tid = threadIdx.x;
  red[tid] = v;
  __syncthreads();
  for (int s = blockDim.x >> 1; s > 0; s >>= 1) {
    if (tid < s) red[tid] += red[tid + s];
    __syncthreads();
  }
  float r = red[0];
  __syncthreads();
  return r;
}

// ---- async global -> LDS 16B copy, tracked by ASYNCcnt --------------------
__device__ __forceinline__ void async_b128(unsigned short* lds, const unsigned short* gp) {
#if __has_builtin(__builtin_amdgcn_global_load_async_to_lds_b128)
  __builtin_amdgcn_global_load_async_to_lds_b128(
      (gptr_b128)(v4i*)const_cast<unsigned short*>(gp),
      (lptr_b128)(v4i*)lds, 0, 0);
#else
  asm volatile("global_load_async_to_lds_b128 %0, %1, off"
               :: "v"((unsigned)(uintptr_t)lds), "v"(gp)
               : "memory");
#endif
}

template <int N>
__device__ __forceinline__ void wait_async() {
#if __has_builtin(__builtin_amdgcn_s_wait_asynccnt)
  __builtin_amdgcn_s_wait_asynccnt((short)N);
#else
  asm volatile("s_wait_asynccnt %0" :: "i"(N) : "memory");
#endif
}

__device__ __forceinline__ v16bf lds_frag(const unsigned short* p, int gap) {
  union { uint4 q[2]; v16bf v; } u;
  u.q[0] = *reinterpret_cast<const uint4*>(p);
  u.q[1] = *reinterpret_cast<const uint4*>(p + gap);
  return u.v;
}

// ---------------- LayerNorm over din=1024, permute group-major, cvt bf16
__global__ void ln_in_kernel(const float* __restrict__ x,
                             const float* __restrict__ gamma,
                             const float* __restrict__ beta,
                             unsigned short* __restrict__ xperm) {
  __shared__ float red[256];
  int n = blockIdx.x;
  int tid = threadIdx.x;
  const float4 v = reinterpret_cast<const float4*>(x + (size_t)n * DIN)[tid];
  float mean = block_reduce_sum(v.x + v.y + v.z + v.w, red) * (1.0f / DIN);
  float dx = v.x - mean, dy = v.y - mean, dz = v.z - mean, dw = v.w - mean;
  float var = block_reduce_sum(dx * dx + dy * dy + dz * dz + dw * dw, red) * (1.0f / DIN);
  float rstd = rsqrtf(var + 1e-6f);
  const float4 g4 = reinterpret_cast<const float4*>(gamma)[tid];
  const float4 b4 = reinterpret_cast<const float4*>(beta)[tid];
  ushort4 o;
  o.x = f2bf(dx * rstd * g4.x + b4.x);
  o.y = f2bf(dy * rstd * g4.y + b4.y);
  o.z = f2bf(dz * rstd * g4.z + b4.z);
  o.w = f2bf(dw * rstd * g4.w + b4.w);
  int g = n & (BB - 1), k = n >> 3;
  reinterpret_cast<ushort4*>(xperm + ((size_t)(g * KPG + k)) * DIN)[tid] = o;
}

// ---------------- LayerNorm over inner=2048 + ReLU, cvt bf16
__global__ void ln2_relu_kernel(const float* __restrict__ hid,
                                const float* __restrict__ gamma,
                                const float* __restrict__ beta,
                                unsigned short* __restrict__ hbf) {
  __shared__ float red[256];
  int r = blockIdx.x;
  int tid = threadIdx.x;
  const float4* row = reinterpret_cast<const float4*>(hid + (size_t)r * INNER);
  float4 v0 = row[2 * tid], v1 = row[2 * tid + 1];
  float s = v0.x + v0.y + v0.z + v0.w + v1.x + v1.y + v1.z + v1.w;
  float mean = block_reduce_sum(s, red) * (1.0f / INNER);
  float d0 = v0.x - mean, d1 = v0.y - mean, d2 = v0.z - mean, d3 = v0.w - mean;
  float d4 = v1.x - mean, d5 = v1.y - mean, d6 = v1.z - mean, d7 = v1.w - mean;
  float sq = d0*d0 + d1*d1 + d2*d2 + d3*d3 + d4*d4 + d5*d5 + d6*d6 + d7*d7;
  float var = block_reduce_sum(sq, red) * (1.0f / INNER);
  float rstd = rsqrtf(var + 1e-6f);
  const float4 g0 = reinterpret_cast<const float4*>(gamma)[2 * tid];
  const float4 g1 = reinterpret_cast<const float4*>(gamma)[2 * tid + 1];
  const float4 b0 = reinterpret_cast<const float4*>(beta)[2 * tid];
  const float4 b1 = reinterpret_cast<const float4*>(beta)[2 * tid + 1];
  ushort4 o0, o1;
  o0.x = f2bf(fmaxf(d0 * rstd * g0.x + b0.x, 0.0f));
  o0.y = f2bf(fmaxf(d1 * rstd * g0.y + b0.y, 0.0f));
  o0.z = f2bf(fmaxf(d2 * rstd * g0.z + b0.z, 0.0f));
  o0.w = f2bf(fmaxf(d3 * rstd * g0.w + b0.w, 0.0f));
  o1.x = f2bf(fmaxf(d4 * rstd * g1.x + b1.x, 0.0f));
  o1.y = f2bf(fmaxf(d5 * rstd * g1.y + b1.y, 0.0f));
  o1.z = f2bf(fmaxf(d6 * rstd * g1.z + b1.z, 0.0f));
  o1.w = f2bf(fmaxf(d7 * rstd * g1.w + b1.w, 0.0f));
  ushort4* orow = reinterpret_cast<ushort4*>(hbf + (size_t)r * INNER);
  orow[2 * tid]     = o0;
  orow[2 * tid + 1] = o1;
}

// ---------------- fp32 [R,C] -> transposed bf16 [C,R] (per batch z)
__global__ void transpose_f32_bf16(const float* __restrict__ in,
                                   unsigned short* __restrict__ out,
                                   int R, int C,
                                   long long inStride, long long outStride) {
  __shared__ float tile[32][33];
  const float* src = in + (size_t)blockIdx.z * inStride;
  unsigned short* dst = out + (size_t)blockIdx.z * outStride;
  int c0 = blockIdx.x * 32, r0 = blockIdx.y * 32;
  int tx = threadIdx.x, ty = threadIdx.y;
#pragma unroll
  for (int i = 0; i < 32; i += 8)
    tile[ty + i][tx] = src[(size_t)(r0 + ty + i) * C + c0 + tx];
  __syncthreads();
#pragma unroll
  for (int i = 0; i < 32; i += 8)
    dst[(size_t)(c0 + ty + i) * R + r0 + tx] = f2bf(tile[tx][ty + i]);
}

// ---------------- async-staged bf16 WMMA GEMM ------------------------------
// Block tile 128(M) x 256(N); 8 waves as 2(M) x 4(N), wave tile 64x64 = 16 WMMA.
// A group-major [8][512][KDIM]; Wt transposed [NDOM][Ntot][KDIM].
// OUTMODE 0: C row = g*512+m.  OUTMODE 1: C row = g + 8*m (scatter to d_out).
template <int KDIM, int OUTMODE>
__global__ __launch_bounds__(256)
void gemm_bf16_wmma(const unsigned short* __restrict__ Abase,
                    const unsigned short* __restrict__ Wt,
                    const float* __restrict__ domain,
                    const float* __restrict__ bias,
                    float* __restrict__ Cout,
                    int Ntot) {
  constexpr int STAGES  = KDIM / 32;
  constexpr int RSTRIDE = 40;               // 32 bf16 + 8 pad -> 80B row stride
  constexpr int ASZ = 128 * RSTRIDE;        // A tile: 128 rows x 32 K
  constexpr int BSZ = 256 * RSTRIDE;        // B tile: 256 cols x 32 K
  constexpr int STAGE = ASZ + BSZ;          // 15360 ushorts = 30 KiB
  __shared__ __align__(16) unsigned short smem[2 * STAGE];   // 60 KiB

  int g = blockIdx.z;
  int d = 0;
#pragma unroll
  for (int j = 1; j < NDOM; ++j)
    if (domain[g * NDOM + j] > 0.5f) d = j;

  const unsigned short* A  = Abase + (size_t)g * KPG * KDIM
                                   + (size_t)blockIdx.y * 128 * KDIM;
  const unsigned short* Bm = Wt + (size_t)d * (size_t)Ntot * KDIM
                                + (size_t)blockIdx.x * 256 * KDIM;

  int tid = threadIdx.x;
  // copy roles: A -> thread covers row tid/2, 32B chunk pair (tid&1)
  int ar = tid >> 1, ac = (tid & 1) * 2;
  const unsigned short* gA = A + (size_t)ar * KDIM + ac * 8;
  // B -> thread covers col tid, full 64B row
  const unsigned short* gB = Bm + (size_t)tid * KDIM;

  int lane = tid & 31, wave = tid >> 5;
  int wm = wave >> 2;                       // 0..1  -> M offset wm*64
  int wn = wave & 3;                        // 0..3  -> N offset wn*64
  int lr = lane & 15, lh = lane >> 4;

  v8f acc[4][4] = {};

  // stage-0 copies
  {
    unsigned short* sb = smem;
    async_b128(sb + ar * RSTRIDE + ac * 8,     gA);
    async_b128(sb + ar * RSTRIDE + ac * 8 + 8, gA + 8);
    unsigned short* lb = sb + ASZ + tid * RSTRIDE;
    async_b128(lb,      gB);
    async_b128(lb + 8,  gB + 8);
    async_b128(lb + 16, gB + 16);
    async_b128(lb + 24, gB + 24);
  }

  for (int s = 0; s < STAGES; ++s) {
    int buf = s & 1;
    if (s) __syncthreads();                 // readers of buf^1 (stage s-1) done
    if (s + 1 < STAGES) {                   // prefetch stage s+1 into buf^1
      int k0 = (s + 1) * 32;
      unsigned short* sb = smem + (buf ^ 1) * STAGE;
      async_b128(sb + ar * RSTRIDE + ac * 8,     gA + k0);
      async_b128(sb + ar * RSTRIDE + ac * 8 + 8, gA + k0 + 8);
      unsigned short* lb = sb + ASZ + tid * RSTRIDE;
      const unsigned short* gb = gB + k0;
      async_b128(lb,      gb);
      async_b128(lb + 8,  gb + 8);
      async_b128(lb + 16, gb + 16);
      async_b128(lb + 24, gb + 24);
      wait_async<6>();                      // in-order: stage-s copies landed
    } else {
      wait_async<0>();
    }
    __syncthreads();                        // all waves' copies visible

    const unsigned short* sa = smem + buf * STAGE;
    const unsigned short* sb = sa + ASZ;
    v16bf a[4], b[4];
#pragma unroll
    for (int i = 0; i < 4; ++i)
      a[i] = lds_frag(sa + (wm * 64 + i * 16 + lr) * RSTRIDE + lh * 8, 16);
#pragma unroll
    for (int j = 0; j < 4; ++j)
      b[j] = lds_frag(sb + (wn * 64 + j * 16 + lr) * RSTRIDE + lh * 8, 16);
#pragma unroll
    for (int i = 0; i < 4; ++i)
#pragma unroll
      for (int j = 0; j < 4; ++j)
        acc[i][j] = __builtin_amdgcn_wmma_f32_16x16x32_bf16(
            false, a[i], false, b[j], (short)0, acc[i][j], false, false);
  }

  int mBlk = blockIdx.y * 128 + wm * 64;
  int nBlk = blockIdx.x * 256 + wn * 64;
#pragma unroll
  for (int j = 0; j < 4; ++j) {
    int nCol = nBlk + j * 16 + lr;
    float bv = bias[(size_t)d * Ntot + nCol];
#pragma unroll
    for (int i = 0; i < 4; ++i) {
#pragma unroll
      for (int r = 0; r < 8; ++r) {
        int m = mBlk + i * 16 + lh * 8 + r;
        size_t rowIdx = (OUTMODE == 0) ? ((size_t)g * KPG + m)
                                       : ((size_t)g + (size_t)BB * m);
        Cout[rowIdx * (size_t)Ntot + nCol] = acc[i][j][r] + bv;
      }
    }
  }
}

extern "C" void kernel_launch(void* const* d_in, const int* in_sizes, int n_in,
                              void* d_out, int out_size, void* d_ws, size_t ws_size,
                              hipStream_t stream) {
  const float* inputs  = (const float*)d_in[0];
  const float* domain  = (const float*)d_in[1];
  const float* ln_g    = (const float*)d_in[2];
  const float* ln_b    = (const float*)d_in[3];
  const float* iln_g   = (const float*)d_in[4];
  const float* iln_b   = (const float*)d_in[5];
  const float* wi      = (const float*)d_in[6];
  const float* bi      = (const float*)d_in[7];
  const float* wo      = (const float*)d_in[8];
  const float* bo      = (const float*)d_in[9];
  float* out = (float*)d_out;

  char* ws = (char*)d_ws;
  size_t off = 0;
  unsigned short* xperm = (unsigned short*)(ws + off); off += (size_t)NTOK * DIN * 2;
  unsigned short* wti   = (unsigned short*)(ws + off); off += (size_t)NDOM * INNER * DIN * 2;
  unsigned short* wto   = (unsigned short*)(ws + off); off += (size_t)NDOM * DOUT * INNER * 2;
  float*          hid   = (float*)(ws + off);          off += (size_t)NTOK * INNER * 4;
  unsigned short* hbf   = (unsigned short*)(ws + off); off += (size_t)NTOK * INNER * 2;
  (void)ws_size; (void)in_sizes; (void)n_in; (void)out_size;

  dim3 tb(32, 8);
  transpose_f32_bf16<<<dim3(NDOM * INNER / 32, DIN / 32, 1), tb, 0, stream>>>(
      wi, wti, DIN, NDOM * INNER, 0, 0);
  transpose_f32_bf16<<<dim3(DOUT / 32, INNER / 32, NDOM), tb, 0, stream>>>(
      wo, wto, INNER, DOUT, (long long)INNER * DOUT, (long long)DOUT * INNER);

  ln_in_kernel<<<NTOK, 256, 0, stream>>>(inputs, ln_g, ln_b, xperm);

  // GEMM1: per group [512,1024] x [1024,2048] -> hid (+inner_bias)
  gemm_bf16_wmma<DIN, 0><<<dim3(INNER / 256, KPG / 128, BB), 256, 0, stream>>>(
      xperm, wti, domain, bi, hid, INNER);

  ln2_relu_kernel<<<NTOK, 256, 0, stream>>>(hid, iln_g, iln_b, hbf);

  // GEMM2: per group [512,2048] x [2048,1024] -> d_out rows n = g + 8*m
  gemm_bf16_wmma<INNER, 1><<<dim3(DOUT / 256, KPG / 128, BB), 256, 0, stream>>>(
      hbf, wto, domain, bo, out, DOUT);
}
